// GaussianPolicy_19533511262907
// MI455X (gfx1250) — compile-verified, matrix operand order checked by reading, not run on hardware
//
#include <hip/hip_runtime.h>

typedef __attribute__((ext_vector_type(16))) __bf16 v16bf;
typedef __attribute__((ext_vector_type(8)))  __bf16 v8bf;
typedef __attribute__((ext_vector_type(4)))  __bf16 v4bf;
typedef __attribute__((ext_vector_type(8)))  float  v8f;
typedef __attribute__((ext_vector_type(4)))  int    v4i;

#define BB 4096
#define SS 5
#define INF 512
#define HH 1024
#define AAD 64

#if __has_builtin(__builtin_amdgcn_global_load_async_to_lds_b128)
#define HAVE_ASYNC_LDS 1
typedef __attribute__((address_space(1))) v4i g_v4i;   // global int4
typedef __attribute__((address_space(3))) v4i l_v4i;   // LDS int4
#endif

// 16-byte global -> LDS stage (async if available, else synchronous copy)
__device__ __forceinline__ void stage16(__bf16* lds, const __bf16* glb) {
#ifdef HAVE_ASYNC_LDS
    v4i* gp0 = (v4i*)glb;      // reinterpret pointee (generic AS)
    v4i* lp0 = (v4i*)lds;
    __builtin_amdgcn_global_load_async_to_lds_b128((g_v4i*)gp0, (l_v4i*)lp0, 0, 0);
#else
    *(v8bf*)lds = *(const v8bf*)glb;
#endif
}

__device__ __forceinline__ void stage_wait() {
#ifdef HAVE_ASYNC_LDS
#if __has_builtin(__builtin_amdgcn_s_wait_asynccnt)
    __builtin_amdgcn_s_wait_asynccnt(0);
#else
    asm volatile("s_wait_asynccnt 0x0" ::: "memory");
#endif
#endif
}

union Frag { v16bf v; v8bf h[2]; };

// ---------------------------------------------------------------------------
// f32 -> bf16 conversion, 4 elements / thread (all sizes divisible by 4)
// ---------------------------------------------------------------------------
__global__ __launch_bounds__(256)
void cvt_f32_bf16_x4(const float4* __restrict__ s, v4bf* __restrict__ d, int n4) {
    int i = blockIdx.x * blockDim.x + threadIdx.x;
    if (i < n4) {
        float4 f = s[i];
        v4bf o;
        o[0] = (__bf16)f.x; o[1] = (__bf16)f.y;
        o[2] = (__bf16)f.z; o[3] = (__bf16)f.w;
        d[i] = o;
    }
}

// ---------------------------------------------------------------------------
// bf16 WMMA GEMM with async-LDS double buffering:
//   C[M,N] = act(A[M,K] * W[N,K]^T + bias[N])
// Wave tile: 32M x 64N (2 A frags x 4 B frags -> acc[2][4], 8 WMMA / K-step).
// Block: WMW x WNW waves -> (WMW*32) x (WNW*64) tile, K-step 32.
// K, N compile-time so all addressing is shifts/adds and fills unroll fully.
// ---------------------------------------------------------------------------
template<int WMW, int WNW, int K, int N, int ACT, bool OUT_F32>
__global__ __launch_bounds__(WMW * WNW * 32)
void wmma_gemm_lds(const __bf16* __restrict__ A, const __bf16* __restrict__ W,
                   const float* __restrict__ bias,
                   float* __restrict__ Cf, __bf16* __restrict__ Cb) {
    constexpr int NTH   = WMW * WNW * 32;  // threads per block
    constexpr int AROWS = WMW * 32;        // M rows per block
    constexpr int BROWS = WNW * 64;        // N cols per block
    constexpr int RPP   = NTH / 4;         // tile rows filled per pass
    constexpr int LDT   = 40;              // padded LDS row stride (bf16), 80B

    __shared__ __bf16 As[2][AROWS * LDT];
    __shared__ __bf16 Bs[2][BROWS * LDT];

    const int tid  = threadIdx.x;
    const int mblk = blockIdx.x * AROWS;
    const int nblk = blockIdx.y * BROWS;

    // cooperative tile fill: each thread owns fixed 16B chunks
    const int crow = tid >> 2;
    const int ccol = (tid & 3) * 8;
    const __bf16* gA = A + (size_t)(mblk + crow) * K + ccol;
    const __bf16* gB = W + (size_t)(nblk + crow) * K + ccol;
    const int lA = crow * LDT + ccol;
    const int lB = crow * LDT + ccol;

    auto stage = [&](int buf, int kk) {
#pragma unroll
        for (int j = 0; j < AROWS / RPP; ++j)
            stage16(&As[buf][lA + j * RPP * LDT], gA + kk + (size_t)j * RPP * K);
#pragma unroll
        for (int j = 0; j < BROWS / RPP; ++j)
            stage16(&Bs[buf][lB + j * RPP * LDT], gB + kk + (size_t)j * RPP * K);
    };

    const int wave = tid >> 5;
    const int lane = tid & 31;
    const int wm   = wave % WMW;
    const int wn   = wave / WMW;
    const int half = lane >> 4;
    const int r    = lane & 15;

    const int aoff = (wm * 32 + r) * LDT + half * 8;   // A frag f adds f*16*LDT
    const int boff = (wn * 64 + r) * LDT + half * 8;   // B frag t adds t*16*LDT

    v8f acc[2][4] = {};

    stage(0, 0);
    int buf = 0;
    for (int kk = 0; kk < K; kk += 32, buf ^= 1) {
        stage_wait();
        __syncthreads();                       // tile[buf] visible to all waves
        if (kk + 32 < K) stage(buf ^ 1, kk + 32);

        Frag a[2], b[4];
#pragma unroll
        for (int f = 0; f < 2; ++f) {
            a[f].h[0] = *(const v8bf*)&As[buf][aoff + f * 16 * LDT];
            a[f].h[1] = *(const v8bf*)&As[buf][aoff + f * 16 * LDT + 16];
        }
#pragma unroll
        for (int t = 0; t < 4; ++t) {
            b[t].h[0] = *(const v8bf*)&Bs[buf][boff + t * 16 * LDT];
            b[t].h[1] = *(const v8bf*)&Bs[buf][boff + t * 16 * LDT + 16];
        }
#pragma unroll
        for (int t = 0; t < 4; ++t)
#pragma unroll
            for (int f = 0; f < 2; ++f)
                acc[f][t] = __builtin_amdgcn_wmma_f32_16x16x32_bf16(
                    false, a[f].v, false, b[t].v, (short)0, acc[f][t], false, false);

        __syncthreads();                       // all reads of tile[buf] done
    }

    // Epilogue. C/D layout: VGPR j -> row m0 + f*16 + half*8 + j, col n0 + t*16 + r.
    const int m0 = mblk + wm * 32;
    const int n0 = nblk + wn * 64;
#pragma unroll
    for (int t = 0; t < 4; ++t) {
        const int col = n0 + t * 16 + r;
        const float bv = bias[col];
#pragma unroll
        for (int f = 0; f < 2; ++f)
#pragma unroll
            for (int j = 0; j < 8; ++j) {
                const int row = m0 + f * 16 + half * 8 + j;
                float v = acc[f][t][j] + bv;
                if (ACT == 1) v = fmaxf(v, 0.0f);
                if (ACT == 2) v = fminf(fmaxf(v, -20.0f), 2.0f);
                if (OUT_F32) Cf[(size_t)row * N + col] = v;
                else         Cb[(size_t)row * N + col] = (__bf16)v;
            }
    }
}

// ---------------------------------------------------------------------------
// Izhikevich scan: proj[B,5,H] f32 -> spike-rate x[B,H] bf16.
// Each of the 5 inputs is applied 3 consecutive steps (repeat(...,3)).
// ---------------------------------------------------------------------------
__global__ __launch_bounds__(256)
void izh_scan_kernel(const float* __restrict__ proj, __bf16* __restrict__ xb) {
    int i = blockIdx.x * blockDim.x + threadIdx.x;   // over B*H
    if (i >= BB * HH) return;
    int b = i >> 10;          // /HH
    int h = i & (HH - 1);     // %HH
    const float* p = proj + (size_t)b * SS * HH + h;

    float v = -65.0f;
    float u = 0.2f * -65.0f;
    float sp = 0.0f;
#pragma unroll
    for (int s = 0; s < SS; ++s) {
        const float I = p[(size_t)s * HH];
#pragma unroll
        for (int r3 = 0; r3 < 3; ++r3) {
            v = v + 0.04f * v * v + 5.0f * v + 140.0f - u + I;
            u = u + 0.02f * (0.2f * v - u);
            if (v >= 30.0f) { v = -65.0f; u += 8.0f; }
            sp += (v - 2.0f > 0.0f) ? 1.0f : 0.0f;
        }
    }
    xb[i] = (__bf16)(sp * (1.0f / 15.0f));
}

// ---------------------------------------------------------------------------
extern "C" void kernel_launch(void* const* d_in, const int* in_sizes, int n_in,
                              void* d_out, int out_size, void* d_ws, size_t ws_size,
                              hipStream_t stream) {
    const float* state = (const float*)d_in[0];
    const float* W_fc  = (const float*)d_in[1];
    const float* b_fc  = (const float*)d_in[2];
    const float* W11   = (const float*)d_in[3];
    const float* b11   = (const float*)d_in[4];
    const float* W12   = (const float*)d_in[5];
    const float* b12   = (const float*)d_in[6];
    const float* W21   = (const float*)d_in[7];
    const float* b21   = (const float*)d_in[8];
    const float* W22   = (const float*)d_in[9];
    const float* b22   = (const float*)d_in[10];
    const float* Wm    = (const float*)d_in[11];
    const float* bm    = (const float*)d_in[12];
    const float* Wls   = (const float*)d_in[13];
    const float* bls   = (const float*)d_in[14];
    float* outp = (float*)d_out;

    char* ws = (char*)d_ws;
    size_t off = 0;
    auto alloc = [&](size_t bytes) -> void* {
        void* p = ws + off;
        off += (bytes + 255) & ~(size_t)255;
        return p;
    };

    __bf16* sbf  = (__bf16*)alloc((size_t)BB * SS * INF * 2);
    __bf16* wfcb = (__bf16*)alloc((size_t)HH * INF * 2);
    __bf16* w11b = (__bf16*)alloc((size_t)HH * HH * 2);
    __bf16* w12b = (__bf16*)alloc((size_t)HH * HH * 2);
    __bf16* w21b = (__bf16*)alloc((size_t)HH * HH * 2);
    __bf16* w22b = (__bf16*)alloc((size_t)HH * HH * 2);
    __bf16* wmb  = (__bf16*)alloc((size_t)AAD * HH * 2);
    __bf16* wlsb = (__bf16*)alloc((size_t)AAD * HH * 2);
    float*  proj = (float*)alloc((size_t)BB * SS * HH * 4);
    __bf16* xb   = (__bf16*)alloc((size_t)BB * HH * 2);
    __bf16* ha   = (__bf16*)alloc((size_t)BB * HH * 2);
    __bf16* hb   = (__bf16*)alloc((size_t)BB * HH * 2);

    auto cvt = [&](const float* src, __bf16* dst, int n) {
        int n4 = n >> 2;
        cvt_f32_bf16_x4<<<(n4 + 255) / 256, 256, 0, stream>>>(
            (const float4*)src, (v4bf*)dst, n4);
    };
    cvt(state, sbf,  BB * SS * INF);
    cvt(W_fc,  wfcb, HH * INF);
    cvt(W11,   w11b, HH * HH);
    cvt(W12,   w12b, HH * HH);
    cvt(W21,   w21b, HH * HH);
    cvt(W22,   w22b, HH * HH);
    cvt(Wm,    wmb,  AAD * HH);
    cvt(Wls,   wlsb, AAD * HH);

    // proj = state @ W_fc^T + b_fc : M=20480, N=1024, K=512 -> f32
    wmma_gemm_lds<2, 4, INF, HH, 0, true>
        <<<dim3((BB * SS) / 64, HH / 256), dim3(256), 0, stream>>>(
        sbf, wfcb, b_fc, proj, nullptr);

    // Izhikevich scan -> x (bf16)
    izh_scan_kernel<<<(BB * HH) / 256, 256, 0, stream>>>(proj, xb);

    // branch 1: x1 = relu(x@W11^T+b11); x1 = relu(x1@W12^T+b12); mean = x1@Wm^T+bm
    wmma_gemm_lds<2, 4, HH, HH, 1, false>
        <<<dim3(BB / 64, HH / 256), dim3(256), 0, stream>>>(
        xb, w11b, b11, nullptr, ha);
    wmma_gemm_lds<2, 4, HH, HH, 1, false>
        <<<dim3(BB / 64, HH / 256), dim3(256), 0, stream>>>(
        ha, w12b, b12, nullptr, hb);
    wmma_gemm_lds<4, 1, HH, AAD, 0, true>
        <<<dim3(BB / 128, AAD / 64), dim3(128), 0, stream>>>(
        hb, wmb, bm, outp, nullptr);

    // branch 2: x2 = relu(x@W21^T+b21); x2 = relu(x2@W22^T+b22);
    // log_std = clip(x2@Wls^T+bls, -20, 2)
    wmma_gemm_lds<2, 4, HH, HH, 1, false>
        <<<dim3(BB / 64, HH / 256), dim3(256), 0, stream>>>(
        xb, w21b, b21, nullptr, ha);
    wmma_gemm_lds<2, 4, HH, HH, 1, false>
        <<<dim3(BB / 64, HH / 256), dim3(256), 0, stream>>>(
        ha, w22b, b22, nullptr, hb);
    wmma_gemm_lds<4, 1, HH, AAD, 2, true>
        <<<dim3(BB / 128, AAD / 64), dim3(128), 0, stream>>>(
        hb, wlsb, bls, outp + (size_t)BB * AAD, nullptr);
}
